// HyenaOperator_15848429322748
// MI455X (gfx1250) — matrix operand order repaired and checked by reading, not compiled, not run on hardware
//
#include <hip/hip_runtime.h>
#include <hip/hip_bf16.h>

typedef _Float16 v16h __attribute__((ext_vector_type(16)));
typedef _Float16 v8h  __attribute__((ext_vector_type(8)));
typedef float    v8f  __attribute__((ext_vector_type(8)));
typedef int      i4v  __attribute__((ext_vector_type(4)));

#define L_SEQ 8192
#define D_DIM 512
#define B_BAT 8
#define KF    64      // surviving frequency bins (H[k] < 1e-14 beyond)
#define KF2   128     // cos+sin rows
#define NSPLIT_B 8    // split-K factor for the DFT stage

#define AS1 __attribute__((address_space(1)))
#define AS3 __attribute__((address_space(3)))

#if defined(__HIP_DEVICE_COMPILE__) && \
    __has_builtin(__builtin_amdgcn_global_load_async_to_lds_b128) && \
    __has_builtin(__builtin_amdgcn_s_wait_asynccnt)
#define HAS_ASYNC 1
#else
#define HAS_ASYNC 0
#endif

// ---------------- small helper kernels ----------------

__global__ void cvt_f32_f16_kernel(const float* __restrict__ src,
                                   _Float16* __restrict__ dst, long long n) {
  long long i = (long long)blockIdx.x * blockDim.x + threadIdx.x;
  long long stride = (long long)gridDim.x * blockDim.x;
  for (; i < n; i += stride) dst[i] = (_Float16)src[i];
}

// W_dft [128][8192]: row k<64 -> cos(2*pi*k*l/L), row 64+k -> sin(...)
__global__ void gen_dft_kernel(_Float16* __restrict__ w) {
  long long idx = (long long)blockIdx.x * blockDim.x + threadIdx.x;
  if (idx >= (long long)KF2 * L_SEQ) return;
  int k = (int)(idx >> 13);
  int l = (int)(idx & (L_SEQ - 1));
  int kk = (k < KF) ? k : (k - KF);
  int phase = (int)(((long long)kk * l) & (L_SEQ - 1));   // k*l mod L
  double ang = (double)phase * (6.283185307179586476925286766559 / (double)L_SEQ);
  double val = (k < KF) ? cos(ang) : sin(ang);
  w[idx] = (_Float16)((float)val);
}

// W_inv [8192][128]: col k<64 -> cos(2*pi*k*l/L), col 64+k -> sin(...)
__global__ void gen_inv_kernel(_Float16* __restrict__ w) {
  long long idx = (long long)blockIdx.x * blockDim.x + threadIdx.x;
  if (idx >= (long long)L_SEQ * KF2) return;
  int l = (int)(idx >> 7);
  int k = (int)(idx & (KF2 - 1));
  int kk = (k < KF) ? k : (k - KF);
  int phase = (int)(((long long)kk * l) & (L_SEQ - 1));
  double ang = (double)phase * (6.283185307179586476925286766559 / (double)L_SEQ);
  double val = (k < KF) ? cos(ang) : sin(ang);
  w[idx] = (_Float16)((float)val);
}

// g_k = w_k * (e^-k + e^-k/2) / L, w_0 = 1 else 2; rows 64.. reuse g_{k-64}
__global__ void gen_scale_kernel(float* __restrict__ s) {
  int t = threadIdx.x;
  if (t >= KF2) return;
  int kk = t & (KF - 1);
  float wk = (kk == 0) ? 1.0f : 2.0f;
  float h = expf(-(float)kk) + expf(-0.5f * (float)kk);
  s[t] = wk * h / (float)L_SEQ;
}

// sum split-K partials, apply filter row scale, convert to f16
// Xp: [B*NSPLIT][D][KF2] f32 ; Xt: [B][D][KF2] f16
__global__ void reduce_split_kernel(const float* __restrict__ Xp,
                                    _Float16* __restrict__ Xt,
                                    const float* __restrict__ rsc) {
  long long idx = (long long)blockIdx.x * blockDim.x + threadIdx.x;
  const long long total = (long long)B_BAT * D_DIM * KF2;
  if (idx >= total) return;
  long long b = idx >> 16;            // / (512*128)
  long long r = idx & 65535;
  int k = (int)(idx & (KF2 - 1));
  float acc = 0.0f;
#pragma unroll
  for (int s = 0; s < NSPLIT_B; s++)
    acc += Xp[(b * NSPLIT_B + s) * 65536 + r];
  Xt[idx] = (_Float16)(acc * rsc[k]);
}

// ---------------- WMMA GEMM ----------------
// C[M,N] = A[M,K_chunk] (f16 row-major, lda) x B (f16 "weight layout" [N][K], ldb)
// z decomposes as (batch, split): b = z/nsplit, s = z%nsplit; k-chunk offset = s*K.
// modes: 0 = f16 row-major out, 1 = f16 transposed out C_t[n][m],
//        2 = f32 row-major out, 3 = f32 transposed out
// optional epilogue: +bias[n], +pos[m*posLd+n]
#define LDSW 40                 // padded LDS row stride in halves, 16B-aligned
#define LDSBUF (128 * LDSW)     // halves per buffer

__global__ __launch_bounds__(256)
void wmma_gemm_kernel(const _Float16* __restrict__ A, long long lda, long long strideA,
                      const _Float16* __restrict__ B, long long ldb, long long strideB,
                      void* __restrict__ Cv, long long ldc, long long strideC,
                      int K, int nsplit, int mode,
                      const float* __restrict__ bias,
                      const float* __restrict__ pos, long long posLd) {
  __shared__ _Float16 sA[3 * LDSBUF];
  __shared__ _Float16 sB[3 * LDSBUF];

  const int tid  = threadIdx.x;
  const int lane = tid & 31;
  const int wid  = tid >> 5;          // 0..7
  const int wm   = wid & 3;           // 4 waves x 32 rows
  const int wn   = wid >> 2;          // 2 waves x 64 cols
  const int bz   = blockIdx.z;
  const int bb   = bz / nsplit;
  const int sp   = bz - bb * nsplit;

  const long long rowM = (long long)blockIdx.y * 128;
  const long long colN = (long long)blockIdx.x * 128;

  const _Float16* Ab = A + (long long)bb * strideA + (long long)sp * K;
  const _Float16* Bb = B + (long long)bb * strideB + (long long)sp * K;

  // per-lane fragment addressing (ISA 16-bit A 16x32 / B 32x16 layouts)
  const int laneN  = lane & 15;
  const int laneHi = (lane >> 4) & 1;
  const int aOff   = (wm * 32 + laneN) * LDSW + (laneHi ? 8 : 0);
  const int bOff   = (wn * 64 + laneN) * LDSW + (laneHi ? 8 : 0);

  v8f acc[2][4];
#pragma unroll
  for (int i = 0; i < 2; i++)
#pragma unroll
    for (int j = 0; j < 4; j++)
#pragma unroll
      for (int e = 0; e < 8; e++) acc[i][j][e] = 0.0f;

  const int nK = K >> 5;               // K / 32
  const int ldr = tid >> 2;            // 0..63
  const int ldc8 = (tid & 3) << 3;     // 0,8,16,24 halves
  const int lOff0 = ldr * LDSW + ldc8;
  const int lOff1 = (ldr + 64) * LDSW + ldc8;

  // persistent per-thread global pointers, advanced 32 halves per issued k-step
  const _Float16* gA0 = Ab + (rowM + ldr) * lda + ldc8;
  const _Float16* gA1 = gA0 + 64 * lda;
  const _Float16* gB0 = Bb + (colN + ldr) * ldb + ldc8;
  const _Float16* gB1 = gB0 + 64 * ldb;
  int issueBase = 0;   // LDS half-offset of the buffer being filled

  auto issue_tile = [&]() {
    _Float16* la0 = &sA[issueBase + lOff0];
    _Float16* la1 = &sA[issueBase + lOff1];
    _Float16* lb0 = &sB[issueBase + lOff0];
    _Float16* lb1 = &sB[issueBase + lOff1];
#if HAS_ASYNC
    __builtin_amdgcn_global_load_async_to_lds_b128(
        (AS1 i4v*)(uintptr_t)gA0, (AS3 i4v*)(uintptr_t)la0, 0, 0);
    __builtin_amdgcn_global_load_async_to_lds_b128(
        (AS1 i4v*)(uintptr_t)gA1, (AS3 i4v*)(uintptr_t)la1, 0, 0);
    __builtin_amdgcn_global_load_async_to_lds_b128(
        (AS1 i4v*)(uintptr_t)gB0, (AS3 i4v*)(uintptr_t)lb0, 0, 0);
    __builtin_amdgcn_global_load_async_to_lds_b128(
        (AS1 i4v*)(uintptr_t)gB1, (AS3 i4v*)(uintptr_t)lb1, 0, 0);
#else
    *(v8h*)la0 = *(const v8h*)gA0;
    *(v8h*)la1 = *(const v8h*)gA1;
    *(v8h*)lb0 = *(const v8h*)gB0;
    *(v8h*)lb1 = *(const v8h*)gB1;
#endif
    gA0 += 32; gA1 += 32; gB0 += 32; gB1 += 32;
    issueBase = (issueBase == 2 * LDSBUF) ? 0 : issueBase + LDSBUF;
  };

  int curBase = 0;     // LDS half-offset of the buffer being consumed
  auto compute_tile = [&]() {
    v16h afrag[2];
#pragma unroll
    for (int mt = 0; mt < 2; mt++) {
      const _Float16* p = &sA[curBase + aOff + mt * 16 * LDSW];
      v8h lo = *(const v8h*)p;
      v8h hi = *(const v8h*)(p + 16);
      afrag[mt] = __builtin_shufflevector(lo, hi, 0, 1, 2, 3, 4, 5, 6, 7,
                                          8, 9, 10, 11, 12, 13, 14, 15);
    }
#pragma unroll
    for (int nt = 0; nt < 4; nt++) {
      const _Float16* q = &sB[curBase + bOff + nt * 16 * LDSW];
      v8h lo = *(const v8h*)q;
      v8h hi = *(const v8h*)(q + 16);
      v16h bfrag = __builtin_shufflevector(lo, hi, 0, 1, 2, 3, 4, 5, 6, 7,
                                           8, 9, 10, 11, 12, 13, 14, 15);
      acc[0][nt] = __builtin_amdgcn_wmma_f32_16x16x32_f16(
          false, afrag[0], false, bfrag, (short)0, acc[0][nt], false, false);
      acc[1][nt] = __builtin_amdgcn_wmma_f32_16x16x32_f16(
          false, afrag[1], false, bfrag, (short)0, acc[1][nt], false, false);
    }
    curBase = (curBase == 2 * LDSBUF) ? 0 : curBase + LDSBUF;
  };

  // prologue: fill pipeline 2 deep (triple-buffered LDS)
  issue_tile();
  if (nK > 1) issue_tile();

  // steady state: unconditional wait(4) (the 4 newer ops belong to k+1)
  for (int ks = 0; ks < nK - 1; ks++) {
#if HAS_ASYNC
    __builtin_amdgcn_s_wait_asynccnt(4);
#endif
    __syncthreads();
    if (ks + 2 < nK) issue_tile();
    compute_tile();
  }
  // last k-step
#if HAS_ASYNC
  __builtin_amdgcn_s_wait_asynccnt(0);
#endif
  __syncthreads();
  compute_tile();

  // epilogue
  _Float16* Ch = (_Float16*)Cv + (long long)bz * strideC;
  float*    Cf = (float*)Cv    + (long long)bz * strideC;

#pragma unroll
  for (int mt = 0; mt < 2; mt++) {
#pragma unroll
    for (int nt = 0; nt < 4; nt++) {
      long long gm0 = rowM + wm * 32 + mt * 16 + laneHi * 8;   // m of r=0
      long long gn  = colN + wn * 64 + nt * 16 + laneN;
      float vals[8];
#pragma unroll
      for (int r = 0; r < 8; r++) vals[r] = acc[mt][nt][r];
      if (bias) {
        float bv = bias[gn];
#pragma unroll
        for (int r = 0; r < 8; r++) vals[r] += bv;
      }
      if (pos) {
#pragma unroll
        for (int r = 0; r < 8; r++) vals[r] += pos[(gm0 + r) * posLd + gn];
      }
      if (mode == 1) {                      // f16 transposed: C_t[n][m], one 16B store
        v8h h;
#pragma unroll
        for (int r = 0; r < 8; r++) h[r] = (_Float16)vals[r];
        *(v8h*)(Ch + gn * ldc + gm0) = h;
      } else if (mode == 0) {               // f16 row-major
#pragma unroll
        for (int r = 0; r < 8; r++) Ch[(gm0 + r) * ldc + gn] = (_Float16)vals[r];
      } else if (mode == 3) {               // f32 transposed (split-K partials)
        float4* p0 = (float4*)(Cf + gn * ldc + gm0);
        p0[0] = make_float4(vals[0], vals[1], vals[2], vals[3]);
        p0[1] = make_float4(vals[4], vals[5], vals[6], vals[7]);
      } else {                              // f32 row-major
#pragma unroll
        for (int r = 0; r < 8; r++) Cf[(gm0 + r) * ldc + gn] = vals[r];
      }
    }
  }
}

// ---------------- host launcher ----------------

extern "C" void kernel_launch(void* const* d_in, const int* in_sizes, int n_in,
                              void* d_out, int out_size, void* d_ws, size_t ws_size,
                              hipStream_t stream) {
  const float* x     = (const float*)d_in[0];   // [8, 8192, 512]
  const float* w_in  = (const float*)d_in[1];   // [512, 512]
  const float* b_in  = (const float*)d_in[2];   // [512]
  const float* w_out = (const float*)d_in[3];   // [512, 512]
  const float* b_out = (const float*)d_in[4];   // [512]
  const float* pos   = (const float*)d_in[5];   // [1, 8192, 512]
  float* out = (float*)d_out;

  char* ws = (char*)d_ws;
  const long long ML   = (long long)B_BAT * L_SEQ;   // 65536
  const long long N_XH = ML * D_DIM;                 // 33,554,432 halves

  _Float16* xh   = (_Float16*)(ws + 0);              // 64 MB (reused: Xp partials, then v)
  _Float16* u_t  = (_Float16*)(ws + 67108864);       // 64 MB  u_t[b][d][l]
  _Float16* X_t  = (_Float16*)(ws + 134217728);      // 1 MB   X_t[b][d][128]
  _Float16* wdft = (_Float16*)(ws + 135266304);      // 2 MB   [128][8192]
  _Float16* winv = (_Float16*)(ws + 137363456);      // 2 MB   [8192][128]
  float*    rsc  = (float*)   (ws + 139460608);      // 512 B
  _Float16* wih  = (_Float16*)(ws + 139461120);      // 0.5 MB
  _Float16* woh  = (_Float16*)(ws + 139985408);      // 0.5 MB
  float*    Xp   = (float*)   (ws + 0);              // 16 MB, aliases dead xh during stage B
  _Float16* v    = xh;                               // aliases xh after reduce

  // conversions + basis generation
  cvt_f32_f16_kernel<<<2048, 256, 0, stream>>>(x, xh, N_XH);
  cvt_f32_f16_kernel<<<256, 256, 0, stream>>>(w_in, wih, (long long)D_DIM * D_DIM);
  cvt_f32_f16_kernel<<<256, 256, 0, stream>>>(w_out, woh, (long long)D_DIM * D_DIM);
  gen_dft_kernel<<<(KF2 * L_SEQ + 255) / 256, 256, 0, stream>>>(wdft);
  gen_inv_kernel<<<(L_SEQ * KF2 + 255) / 256, 256, 0, stream>>>(winv);
  gen_scale_kernel<<<1, 128, 0, stream>>>(rsc);

  // Stage A: u = x @ w_in^T + b_in + pos_emb  -> u_t[b][d][l] (f16 transposed)
  wmma_gemm_kernel<<<dim3(D_DIM / 128, L_SEQ / 128, B_BAT), 256, 0, stream>>>(
      xh, D_DIM, (long long)L_SEQ * D_DIM,
      wih, D_DIM, 0,
      u_t, L_SEQ, (long long)D_DIM * L_SEQ,
      D_DIM, 1, /*mode=*/1, b_in, pos, D_DIM);

  // Stage B: X[k,d] = W_dft[k,:] . u[:,d]  (split-K by 8 -> f32 partials, transposed)
  wmma_gemm_kernel<<<dim3(D_DIM / 128, KF2 / 128, B_BAT * NSPLIT_B), 256, 0, stream>>>(
      wdft, L_SEQ, 0,
      u_t, L_SEQ, (long long)D_DIM * L_SEQ,
      Xp, KF2, (long long)D_DIM * KF2,
      L_SEQ / NSPLIT_B, NSPLIT_B, /*mode=*/3, nullptr, nullptr, 0);

  // reduce partials, apply g_k, convert -> X_t f16
  reduce_split_kernel<<<(B_BAT * D_DIM * KF2 + 255) / 256, 256, 0, stream>>>(Xp, X_t, rsc);

  // Stage C: v[l,d] = W_inv[l,:] . Xg[:,d]  -> v row-major f16
  wmma_gemm_kernel<<<dim3(D_DIM / 128, L_SEQ / 128, B_BAT), 256, 0, stream>>>(
      winv, KF2, 0,
      X_t, KF2, (long long)D_DIM * KF2,
      v, D_DIM, (long long)L_SEQ * D_DIM,
      KF2, 1, /*mode=*/0, nullptr, nullptr, 0);

  // Stage D: out = v @ w_out^T + b_out  (f32 out), flat M = 65536
  wmma_gemm_kernel<<<dim3(D_DIM / 128, (unsigned)(ML / 128), 1), 256, 0, stream>>>(
      v, D_DIM, 0,
      woh, D_DIM, 0,
      out, D_DIM, 0,
      D_DIM, 1, /*mode=*/2, b_out, nullptr, 0);
}
